// Critic_without_recurrent_86483461472854
// MI455X (gfx1250) — matrix operand order repaired and checked by reading, not compiled, not run on hardware
//
#include <hip/hip_runtime.h>

// ---------------------------------------------------------------------------
// Spiking MLP (IF x3 + non-spiking LIF head), T=16, batch 4096.
//   dv1 = x@W1^T+b1 computed ONCE (loop invariant).
//   Per step: IF(layer1 elementwise) -> GEMM+IF(fused) x2 -> GEMM+LIF(fused).
//   Spikes are {0,1} -> exact in f16; GEMMs use v_wmma_f32_16x16x32_f16.
// ---------------------------------------------------------------------------

typedef __attribute__((ext_vector_type(16))) _Float16 v16h;
typedef __attribute__((ext_vector_type(8)))  _Float16 v8h;
typedef __attribute__((ext_vector_type(4)))  _Float16 v4h;
typedef __attribute__((ext_vector_type(8)))  float    v8f;

static constexpr int BATCH  = 4096;
static constexpr int D_IN   = 512;
static constexpr int HID    = 1024;
static constexpr int ACT    = 64;
static constexpr int TSTEPS = 16;

union AF { v16h v; v8h h[2]; };   // A fragment (16x32 f16)
union CF { v8f  v; float e[8]; }; // C/D fragment (16x16 f32)

// ---------------- elementwise helper kernels (float4-vectorized) -----------

__global__ __launch_bounds__(256)
void cvt_f32_to_f16(const float* __restrict__ in, _Float16* __restrict__ out, int n4) {
  int i = blockIdx.x * blockDim.x + threadIdx.x;
  if (i >= n4) return;
  float4 f = ((const float4*)in)[i];
  v4h h = { (_Float16)f.x, (_Float16)f.y, (_Float16)f.z, (_Float16)f.w };
  ((v4h*)out)[i] = h;
}

__global__ __launch_bounds__(256)
void fill_zero_f32(float* __restrict__ p, int n4) {
  int i = blockIdx.x * blockDim.x + threadIdx.x;
  if (i >= n4) return;
  ((float4*)p)[i] = make_float4(0.f, 0.f, 0.f, 0.f);
}

// Layer-1 IF step: v += dv; s = (v >= 1); v *= (1 - s)   (dv is loop-invariant)
__global__ __launch_bounds__(256)
void if_layer1(const float* __restrict__ dv, float* __restrict__ v,
               _Float16* __restrict__ s, int n4) {
  int i = blockIdx.x * blockDim.x + threadIdx.x;
  if (i >= n4) return;
  float4 d  = ((const float4*)dv)[i];
  float4 vv = ((float4*)v)[i];
  float a0 = vv.x + d.x, a1 = vv.y + d.y, a2 = vv.z + d.z, a3 = vv.w + d.w;
  float s0 = (a0 >= 1.f) ? 1.f : 0.f;
  float s1 = (a1 >= 1.f) ? 1.f : 0.f;
  float s2 = (a2 >= 1.f) ? 1.f : 0.f;
  float s3 = (a3 >= 1.f) ? 1.f : 0.f;
  ((float4*)v)[i] = make_float4(a0 * (1.f - s0), a1 * (1.f - s1),
                                a2 * (1.f - s2), a3 * (1.f - s3));
  v4h sh = { (_Float16)s0, (_Float16)s1, (_Float16)s2, (_Float16)s3 };
  ((v4h*)s)[i] = sh;
}

// ---------------- fused WMMA GEMM + neuron update --------------------------
// C[M,N] = S[M,K] (f16 row-major) @ W[N,K]^T (f16 row-major) + bias[N]
// Grid: x = N/NT tiles, y = BATCH/128 tiles. 256 threads = 8 wave32s,
// arranged WM x WN; wave tile = (128/WM) x (NT/WN).
// MODE 0: V = dv           (prologue dv1)
// MODE 1: IF update on V, write spike f16 to Sout
// MODE 2: LIF: V += (dv - V) * 0.5   (tau = 2)
template<int K, int N, int NT, int WM, int WN, int MODE>
__global__ __launch_bounds__(256)
void snn_gemm(const _Float16* __restrict__ S,
              const _Float16* __restrict__ W,
              const float*    __restrict__ bias,
              float*          __restrict__ V,
              _Float16*       __restrict__ Sout)
{
  constexpr int FM = 128 / (WM * 16);
  constexpr int FN = NT  / (WN * 16);

  const int lane  = threadIdx.x & 31;
  const int wave  = threadIdx.x >> 5;
  const int wm    = wave % WM;
  const int wn    = wave / WM;
  const int tileM = blockIdx.y * 128;
  const int tileN = blockIdx.x * NT;
  const int l15   = lane & 15;
  const int hiA   = (lane >> 4) << 4;   // +16 halves for upper half-wave (A)
  const int hiB   = (lane >> 4) << 4;   // +16 halves for upper half-wave (B)

  const int mA = tileM + wm * (FM * 16) + l15;  // A-matrix row for this lane
  const int nB = tileN + wn * (FN * 16) + l15;  // B-matrix col for this lane

  v8f acc[FM][FN] = {};

  for (int k0 = 0; k0 < K; k0 += 32) {
    AF a[FM];
#pragma unroll
    for (int i = 0; i < FM; ++i) {
      const _Float16* p = S + (size_t)(mA + i * 16) * K + k0 + hiA;
      a[i].h[0] = *(const v8h*)(p);        // K = base .. base+7
      a[i].h[1] = *(const v8h*)(p + 8);    // K = base+8 .. base+15
    }
    v16h b[FN];
#pragma unroll
    for (int j = 0; j < FN; ++j) {
      // column n of W^T == row n of W: contiguous 32B per lane
      const _Float16* p = W + (size_t)(nB + j * 16) * K + k0 + hiB;
      b[j] = *(const v16h*)(p);
    }
#pragma unroll
    for (int i = 0; i < FM; ++i)
#pragma unroll
      for (int j = 0; j < FN; ++j)
        acc[i][j] = __builtin_amdgcn_wmma_f32_16x16x32_f16(
            false, a[i].v, false, b[j], (short)0, acc[i][j], false, false);
  }

  const int mRow0 = tileM + wm * (FM * 16) + ((lane >> 4) << 3);
#pragma unroll
  for (int i = 0; i < FM; ++i) {
#pragma unroll
    for (int j = 0; j < FN; ++j) {
      const int   n  = nB + j * 16;
      const float bn = bias[n];
      CF c; c.v = acc[i][j];
#pragma unroll
      for (int r = 0; r < 8; ++r) {
        const int    m   = mRow0 + i * 16 + r;
        const size_t idx = (size_t)m * N + n;
        const float  dv  = c.e[r] + bn;
        if constexpr (MODE == 0) {
          V[idx] = dv;
        } else if constexpr (MODE == 1) {
          float vm = V[idx] + dv;
          const float sp = (vm >= 1.0f) ? 1.0f : 0.0f;
          V[idx]    = vm * (1.0f - sp);
          Sout[idx] = (_Float16)sp;
        } else {
          const float vm = V[idx];
          V[idx] = vm + (dv - vm) * 0.5f;
        }
      }
    }
  }
}

// ---------------------------------------------------------------------------

extern "C" void kernel_launch(void* const* d_in, const int* in_sizes, int n_in,
                              void* d_out, int out_size, void* d_ws, size_t ws_size,
                              hipStream_t stream)
{
  (void)in_sizes; (void)n_in; (void)out_size; (void)ws_size;

  const float* x  = (const float*)d_in[0];
  const float* W1 = (const float*)d_in[1];
  const float* b1 = (const float*)d_in[2];
  const float* W2 = (const float*)d_in[3];
  const float* b2 = (const float*)d_in[4];
  const float* W3 = (const float*)d_in[5];
  const float* b3 = (const float*)d_in[6];
  const float* W4 = (const float*)d_in[7];
  const float* b4 = (const float*)d_in[8];

  char* ws = (char*)d_ws;
  size_t off = 0;
  auto take = [&](size_t bytes) -> void* {
    void* p = ws + off;
    off += (bytes + 255) & ~(size_t)255;
    return p;
  };

  _Float16* xh  = (_Float16*)take((size_t)BATCH * D_IN * 2);
  _Float16* W1h = (_Float16*)take((size_t)HID * D_IN * 2);
  _Float16* W2h = (_Float16*)take((size_t)HID * HID * 2);
  _Float16* W3h = (_Float16*)take((size_t)HID * HID * 2);
  _Float16* W4h = (_Float16*)take((size_t)ACT * HID * 2);
  float* dv1 = (float*)take((size_t)BATCH * HID * 4);
  float* v1  = (float*)take((size_t)BATCH * HID * 4);
  float* v2  = (float*)take((size_t)BATCH * HID * 4);
  float* v3  = (float*)take((size_t)BATCH * HID * 4);
  _Float16* s1 = (_Float16*)take((size_t)BATCH * HID * 2);
  _Float16* s2 = (_Float16*)take((size_t)BATCH * HID * 2);
  _Float16* s3 = (_Float16*)take((size_t)BATCH * HID * 2);
  float* v4 = (float*)d_out;   // 4096 x 64, accumulated in place

  auto cvt = [&](const float* in, _Float16* out, int n) {
    int n4 = n / 4;
    cvt_f32_to_f16<<<(n4 + 255) / 256, 256, 0, stream>>>(in, out, n4);
  };
  cvt(x,  xh,  BATCH * D_IN);
  cvt(W1, W1h, HID * D_IN);
  cvt(W2, W2h, HID * HID);
  cvt(W3, W3h, HID * HID);
  cvt(W4, W4h, ACT * HID);

  auto zero = [&](float* p, int n) {
    int n4 = n / 4;
    fill_zero_f32<<<(n4 + 255) / 256, 256, 0, stream>>>(p, n4);
  };
  zero(v1, BATCH * HID);
  zero(v2, BATCH * HID);
  zero(v3, BATCH * HID);
  zero(v4, BATCH * ACT);

  dim3 blk(256);
  dim3 gH(HID / 128, BATCH / 128);  // 8 x 32 tiles for N=1024 layers
  dim3 gA(ACT / 64,  BATCH / 128);  // 1 x 32 tiles for N=64 head

  // dv1 = x @ W1^T + b1 : loop-invariant, computed once
  snn_gemm<D_IN, HID, 128, 4, 2, 0><<<gH, blk, 0, stream>>>(xh, W1h, b1, dv1, nullptr);

  const int n4h = BATCH * HID / 4;
  for (int t = 0; t < TSTEPS; ++t) {
    if_layer1<<<(n4h + 255) / 256, 256, 0, stream>>>(dv1, v1, s1, n4h);
    snn_gemm<HID, HID, 128, 4, 2, 1><<<gH, blk, 0, stream>>>(s1, W2h, b2, v2, s2);
    snn_gemm<HID, HID, 128, 4, 2, 1><<<gH, blk, 0, stream>>>(s2, W3h, b3, v3, s3);
    snn_gemm<HID, ACT, 64, 8, 1, 2><<<gA, blk, 0, stream>>>(s3, W4h, b4, v4, nullptr);
  }
}
